// SparseSelfAttention_19971597926681
// MI455X (gfx1250) — matrix operand order repaired and checked by reading, not compiled, not run on hardware
//
#include <hip/hip_runtime.h>

// ---------------------------------------------------------------------------
// Types for CDNA5 WMMA (wave32): v_wmma_f32_16x16x32_bf16
// ---------------------------------------------------------------------------
typedef __attribute__((ext_vector_type(16))) __bf16 v16bf;
typedef __attribute__((ext_vector_type(8)))  float  v8f;
typedef __attribute__((ext_vector_type(4)))  unsigned u32x4;
typedef __attribute__((ext_vector_type(8)))  unsigned u32x8;

union V16 {
    v16bf v;
    __bf16 e[16];
    unsigned short u[16];
};

__device__ __forceinline__ v8f wmma_bf16(v16bf a, v16bf b, v8f c) {
    return __builtin_amdgcn_wmma_f32_16x16x32_bf16(false, a, false, b,
                                                   (short)0, c, false, false);
}

__device__ __forceinline__ v16bf load16(const __bf16* p) {
    V16 t;
    const uint4* q = reinterpret_cast<const uint4*>(p);
    *reinterpret_cast<uint4*>(&t.u[0]) = q[0];
    *reinterpret_cast<uint4*>(&t.u[8]) = q[1];
    return t.v;
}

// ---------------------------------------------------------------------------
// CDNA5 Tensor Data Mover: 2D tile (global -> LDS), D# built in SGPRs.
//   Loads a tile of tile_y rows x tile_x bf16 elements (row stride
//   stride_elems in the source tensor) into LDS at ldsoff, packed
//   contiguously (tile_x elements per row). Tracked by TENSORcnt.
// ---------------------------------------------------------------------------
__device__ __forceinline__ void tdm_load_tile_bf16(
    const __bf16* gtile, unsigned ldsoff,
    unsigned tile_x, unsigned tile_y, unsigned stride_elems) {
    unsigned long long ga = (unsigned long long)(uintptr_t)gtile;
    u32x4 g0;
    g0[0] = 1u;                                    // count=1, user descriptor
    g0[1] = ldsoff;                                // lds_addr (bytes)
    g0[2] = (unsigned)(ga & 0xffffffffu);          // global_addr[31:0]
    g0[3] = (unsigned)((ga >> 32) & 0x01ffffffu)   // global_addr[56:32]
          | (2u << 30);                            // type = 2 ("image")
    u32x8 g1;
    g1[0] = (1u << 16);                            // data_size=1 -> 2 bytes
    g1[1] = (tile_x & 0xffffu) << 16;              // tensor_dim0[15:0]
    g1[2] = (tile_x >> 16) | ((tile_y & 0xffffu) << 16);   // dim0 hi | dim1 lo
    g1[3] = (tile_y >> 16) | (tile_x << 16);       // dim1 hi | tile_dim0
    g1[4] = tile_y & 0xffffu;                      // tile_dim1 (tile_dim2=0)
    g1[5] = stride_elems;                          // tensor_dim0_stride[31:0]
    g1[6] = 0u;                                    // stride hi, dim1_stride lo
    g1[7] = 0u;
    asm volatile("tensor_load_to_lds %0, %1"
                 :: "s"(g0), "s"(g1)
                 : "memory");
}

__device__ __forceinline__ void wait_tensor0() {
    __builtin_amdgcn_s_wait_tensorcnt((short)0);
}

// monotone map: f32 -> u32 preserving order (for radix top-k select)
__device__ __forceinline__ unsigned mono(float f) {
    unsigned b = __float_as_uint(f);
    return (b & 0x80000000u) ? ~b : (b | 0x80000000u);
}

#define SEQ    2048
#define HDIM   4096
#define NHEADS 32
#define DHEAD  128
#define TOPK   409

// ---------------------------------------------------------------------------
// f32 -> bf16 conversion
// ---------------------------------------------------------------------------
__global__ void cvt_f32_bf16(const float* __restrict__ in,
                             __bf16* __restrict__ out, int n) {
    int i = (blockIdx.x * blockDim.x + threadIdx.x) * 4;
    if (i + 3 < n) {
        float4 v = *reinterpret_cast<const float4*>(in + i);
        out[i + 0] = (__bf16)v.x;
        out[i + 1] = (__bf16)v.y;
        out[i + 2] = (__bf16)v.z;
        out[i + 3] = (__bf16)v.w;
    } else {
        for (int j = i; j < n; ++j) out[j] = (__bf16)in[j];
    }
}

// ---------------------------------------------------------------------------
// GEMM: C[m,n] = sum_k A[m,k] * W[n,k] + bias[n]
//   A: [M,K] bf16 row-major, W: [N,K] bf16 row-major
//   Block tile 128x128, BK=32, 8 waves; double-buffered LDS staged by the
//   Tensor Data Mover (wave0: A tile, wave1: W tile), WMMAs back-to-back.
//   out_mode: 0 = bf16 [M,N], 1 = f32 [M,N], 2 = bf16 transposed [N,M]
// ---------------------------------------------------------------------------
#define BM 128
#define BN 128
#define BK 32

__global__ __launch_bounds__(256) void gemm_bf16(
    const __bf16* __restrict__ A, const __bf16* __restrict__ W,
    const float* __restrict__ bias, void* __restrict__ Cout,
    int M, int N, int K, int out_mode) {

    __shared__ __align__(16) __bf16 As[2][BM][BK];
    __shared__ __align__(16) __bf16 Ws[2][BN][BK];

    const int tid  = threadIdx.x;
    const int lane = tid & 31;
    const int wv   = tid >> 5;          // 0..7
    const int half = lane >> 4;         // 0/1
    const int l15  = lane & 15;

    const int m0 = blockIdx.y * BM;
    const int n0 = blockIdx.x * BN;

    v8f zero = {};
    v8f acc[8];
#pragma unroll
    for (int j = 0; j < 8; ++j) acc[j] = zero;

    const int nk = K / BK;
    int buf = 0;

    // TDM prefetch of stage 0 (wave0 -> A tile, wave1 -> W tile)
    if (wv == 0)
        tdm_load_tile_bf16(A + (size_t)m0 * K, (unsigned)(uintptr_t)&As[0][0][0],
                           BK, BM, (unsigned)K);
    if (wv == 1)
        tdm_load_tile_bf16(W + (size_t)n0 * K, (unsigned)(uintptr_t)&Ws[0][0][0],
                           BK, BN, (unsigned)K);
    if (wv < 2) wait_tensor0();
    __syncthreads();

    for (int kt = 0; kt < nk - 1; ++kt) {
        // TDM prefetch of next K-tile overlaps WMMA on current tile
        const int k1 = (kt + 1) * BK;
        if (wv == 0)
            tdm_load_tile_bf16(A + (size_t)m0 * K + k1,
                               (unsigned)(uintptr_t)&As[buf ^ 1][0][0],
                               BK, BM, (unsigned)K);
        if (wv == 1)
            tdm_load_tile_bf16(W + (size_t)n0 * K + k1,
                               (unsigned)(uintptr_t)&Ws[buf ^ 1][0][0],
                               BK, BN, (unsigned)K);

        // A fragment for this wave's 16-row strip (16-bit A layout 16x32)
        V16 a;
        {
            const int m = wv * 16 + l15;
            *reinterpret_cast<uint4*>(&a.u[0]) =
                *reinterpret_cast<const uint4*>(&As[buf][m][half * 16]);
            *reinterpret_cast<uint4*>(&a.u[8]) =
                *reinterpret_cast<const uint4*>(&As[buf][m][half * 16 + 8]);
        }
        V16 b[8];
#pragma unroll
        for (int j = 0; j < 8; ++j)
            b[j].v = load16(&Ws[buf][j * 16 + l15][half * 16]);
#pragma unroll
        for (int j = 0; j < 8; ++j)
            acc[j] = wmma_bf16(a.v, b[j].v, acc[j]);

        if (wv < 2) wait_tensor0();
        __syncthreads();
        buf ^= 1;
    }

    // last K-tile (no prefetch)
    {
        V16 a;
        {
            const int m = wv * 16 + l15;
            *reinterpret_cast<uint4*>(&a.u[0]) =
                *reinterpret_cast<const uint4*>(&As[buf][m][half * 16]);
            *reinterpret_cast<uint4*>(&a.u[8]) =
                *reinterpret_cast<const uint4*>(&As[buf][m][half * 16 + 8]);
        }
        V16 b[8];
#pragma unroll
        for (int j = 0; j < 8; ++j)
            b[j].v = load16(&Ws[buf][j * 16 + l15][half * 16]);
#pragma unroll
        for (int j = 0; j < 8; ++j)
            acc[j] = wmma_bf16(a.v, b[j].v, acc[j]);
    }

    // epilogue: D layout — VGPR r: lanes0-15 -> M=r, lanes16-31 -> M=r+8
#pragma unroll
    for (int j = 0; j < 8; ++j) {
        const int col = n0 + j * 16 + l15;
        const float bv = bias[col];
#pragma unroll
        for (int r = 0; r < 8; ++r) {
            const int row = m0 + wv * 16 + half * 8 + r;
            const float val = acc[j][r] + bv;
            if (out_mode == 1) {
                reinterpret_cast<float*>(Cout)[(size_t)row * N + col] = val;
            } else if (out_mode == 0) {
                reinterpret_cast<__bf16*>(Cout)[(size_t)row * N + col] = (__bf16)val;
            } else {  // transposed bf16: Vt[col][row], minor dim = M
                reinterpret_cast<__bf16*>(Cout)[(size_t)col * M + row] = (__bf16)val;
            }
        }
    }
}

// ---------------------------------------------------------------------------
// Sparse attention for one (head, 16-query tile):
//   scores = Q Kt / sqrt(d)  (WMMA)  -> LDS f32 [16][2048]
//   exact top-k threshold per row via 4-pass radix select (LDS histogram)
//   masked softmax -> P bf16 [16][2048] in LDS
//   ctx = P V (WMMA, V pre-transposed to Vt[d][key]) -> bf16 [2048][4096]
// ---------------------------------------------------------------------------
__global__ __launch_bounds__(256) void attn_topk(
    const __bf16* __restrict__ Q, const __bf16* __restrict__ Km,
    const __bf16* __restrict__ Vt, __bf16* __restrict__ Ctx,
    float scale) {

    extern __shared__ char smem[];
    float*    Sc   = reinterpret_cast<float*>(smem);                       // 16*2048 f32
    __bf16*   P    = reinterpret_cast<__bf16*>(smem + 16 * SEQ * 4);       // 16*2048 bf16
    unsigned* hist = reinterpret_cast<unsigned*>(smem + 16 * SEQ * 4 + 16 * SEQ * 2);
    unsigned* thru = hist + 256;                                           // 16
    unsigned* sel  = thru + 16;                                            // 2

    const int tid  = threadIdx.x;
    const int lane = tid & 31;
    const int wv   = tid >> 5;      // 0..7
    const int half = lane >> 4;
    const int l15  = lane & 15;

    const int qt = blockIdx.x;      // 0..127
    const int h  = blockIdx.y;      // 0..31
    const int qbase = qt * 16;
    const size_t hoff = (size_t)h * DHEAD;

    // ---- pre-load the four Q A-fragments (d = 0..127 in 4 chunks of 32) ----
    V16 qa[4];
    {
        const __bf16* qrow = Q + (size_t)(qbase + l15) * HDIM + hoff;
#pragma unroll
        for (int dk = 0; dk < 4; ++dk) {
            *reinterpret_cast<uint4*>(&qa[dk].u[0]) =
                *reinterpret_cast<const uint4*>(qrow + dk * 32 + half * 16);
            *reinterpret_cast<uint4*>(&qa[dk].u[8]) =
                *reinterpret_cast<const uint4*>(qrow + dk * 32 + half * 16 + 8);
        }
    }

    // ---- scores: wave wv handles key tiles kt = wv, wv+8, ... (16 tiles) ----
    v8f zero = {};
    for (int t = 0; t < 16; ++t) {
        const int kt = wv + t * 8;
        v8f acc = zero;
        V16 b[4];
#pragma unroll
        for (int dk = 0; dk < 4; ++dk)
            b[dk].v = load16(Km + (size_t)(kt * 16 + l15) * HDIM + hoff +
                             dk * 32 + half * 16);
#pragma unroll
        for (int dk = 0; dk < 4; ++dk)
            acc = wmma_bf16(qa[dk].v, b[dk].v, acc);
#pragma unroll
        for (int r = 0; r < 8; ++r) {
            const int row = half * 8 + r;
            const int col = kt * 16 + l15;
            Sc[row * SEQ + col] = acc[r] * scale;
        }
    }
    __syncthreads();

    // ---- exact top-k threshold per row: 4-pass radix select on mono(u32) ----
    for (int r = 0; r < 16; ++r) {
        if (tid == 0) { sel[0] = 0u; sel[1] = (unsigned)TOPK; }
        __syncthreads();
        for (int pass = 0; pass < 4; ++pass) {
            const int shift = 24 - pass * 8;
            hist[tid] = 0u;
            __syncthreads();
            const unsigned pref = sel[0];
            for (int e = tid; e < SEQ; e += 256) {
                unsigned u = mono(Sc[r * SEQ + e]);
                bool match = (pass == 0) ||
                             ((u >> (shift + 8)) == (pref >> (shift + 8)));
                if (match) atomicAdd(&hist[(u >> shift) & 255u], 1u);
            }
            __syncthreads();
            if (tid == 0) {
                unsigned rem = sel[1];
                unsigned p2  = pref;
                for (int b = 255; b >= 0; --b) {
                    unsigned c = hist[b];
                    if (c >= rem) { p2 |= ((unsigned)b) << shift; break; }
                    rem -= c;
                }
                sel[0] = p2; sel[1] = rem;
            }
            __syncthreads();
        }
        if (tid == 0) thru[r] = sel[0];   // exact value of k-th largest
        __syncthreads();
    }

    // ---- masked softmax: wave wv handles rows 2*wv and 2*wv+1 ----
    for (int rr = 0; rr < 2; ++rr) {
        const int r = wv * 2 + rr;
        const unsigned tu = thru[r];
        float mx = -3.4e38f;
        for (int e = lane; e < SEQ; e += 32) {
            float s = Sc[r * SEQ + e];
            if (mono(s) >= tu) mx = fmaxf(mx, s);
        }
#pragma unroll
        for (int o = 16; o > 0; o >>= 1) mx = fmaxf(mx, __shfl_xor(mx, o, 32));
        float sm = 0.f;
        for (int e = lane; e < SEQ; e += 32) {
            float s = Sc[r * SEQ + e];
            float ev = (mono(s) >= tu) ? __expf(s - mx) : 0.f;
            Sc[r * SEQ + e] = ev;       // store exp once, reuse for normalize
            sm += ev;
        }
#pragma unroll
        for (int o = 16; o > 0; o >>= 1) sm += __shfl_xor(sm, o, 32);
        const float inv = 1.f / sm;
        for (int e = lane; e < SEQ; e += 32) {
            P[r * SEQ + e] = (__bf16)(Sc[r * SEQ + e] * inv);
        }
    }
    __syncthreads();

    // ---- ctx = P * V : wave wv owns output cols d0 = wv*16 ----
    {
        const int d0 = wv * 16;
        v8f acc = zero;
        for (int kk = 0; kk < SEQ; kk += 32) {
            V16 a;
            const __bf16* pp = P + (size_t)l15 * SEQ + kk;
            *reinterpret_cast<uint4*>(&a.u[0]) =
                *reinterpret_cast<const uint4*>(pp + half * 16);
            *reinterpret_cast<uint4*>(&a.u[8]) =
                *reinterpret_cast<const uint4*>(pp + half * 16 + 8);
            v16bf b = load16(Vt + (hoff + d0 + l15) * (size_t)SEQ + kk + half * 16);
            acc = wmma_bf16(a.v, b, acc);
        }
#pragma unroll
        for (int r = 0; r < 8; ++r) {
            const int qrow = qbase + half * 8 + r;
            Ctx[(size_t)qrow * HDIM + hoff + d0 + l15] = (__bf16)acc[r];
        }
    }
}

// ---------------------------------------------------------------------------
// launch
// ---------------------------------------------------------------------------
extern "C" void kernel_launch(void* const* d_in, const int* in_sizes, int n_in,
                              void* d_out, int out_size, void* d_ws, size_t ws_size,
                              hipStream_t stream) {
    (void)in_sizes; (void)n_in; (void)out_size; (void)ws_size;

    const float* x  = (const float*)d_in[0];
    const float* wq = (const float*)d_in[1];
    const float* bq = (const float*)d_in[2];
    const float* wk = (const float*)d_in[3];
    const float* bk = (const float*)d_in[4];
    const float* wv_ = (const float*)d_in[5];
    const float* bv = (const float*)d_in[6];
    const float* wo = (const float*)d_in[7];
    const float* bo = (const float*)d_in[8];

    char* ws = (char*)d_ws;
    size_t off = 0;
    auto carve = [&](size_t bytes) -> void* {
        void* p = ws + off;
        off += (bytes + 255) & ~(size_t)255;
        return p;
    };
    const size_t actB = (size_t)SEQ * HDIM * sizeof(__bf16);    // 16 MB
    const size_t wgtB = (size_t)HDIM * HDIM * sizeof(__bf16);   // 32 MB

    __bf16* xb  = (__bf16*)carve(actB);
    __bf16* wqb = (__bf16*)carve(wgtB);
    __bf16* wkb = (__bf16*)carve(wgtB);
    __bf16* wvb = (__bf16*)carve(wgtB);
    __bf16* wob = (__bf16*)carve(wgtB);
    __bf16* Qb  = (__bf16*)carve(actB);
    __bf16* Kb  = (__bf16*)carve(actB);
    __bf16* Vtb = (__bf16*)carve(actB);   // transposed: Vt[d][key]
    __bf16* Cx  = (__bf16*)carve(actB);

    // 1) convert to bf16
    {
        int nA = SEQ * HDIM;
        int nW = HDIM * HDIM;
        cvt_f32_bf16<<<(nA / 4 + 255) / 256, 256, 0, stream>>>(x, xb, nA);
        cvt_f32_bf16<<<(nW / 4 + 255) / 256, 256, 0, stream>>>(wq, wqb, nW);
        cvt_f32_bf16<<<(nW / 4 + 255) / 256, 256, 0, stream>>>(wk, wkb, nW);
        cvt_f32_bf16<<<(nW / 4 + 255) / 256, 256, 0, stream>>>(wv_, wvb, nW);
        cvt_f32_bf16<<<(nW / 4 + 255) / 256, 256, 0, stream>>>(wo, wob, nW);
    }

    // 2) Q/K/V projections (bf16 out; V written transposed for P·V)
    {
        dim3 grid(HDIM / BN, SEQ / BM);
        gemm_bf16<<<grid, 256, 0, stream>>>(xb, wqb, bq, Qb, SEQ, HDIM, HDIM, 0);
        gemm_bf16<<<grid, 256, 0, stream>>>(xb, wkb, bk, Kb, SEQ, HDIM, HDIM, 0);
        gemm_bf16<<<grid, 256, 0, stream>>>(xb, wvb, bv, Vtb, SEQ, HDIM, HDIM, 2);
    }

    // 3) top-k sparse attention (per head, per 16-query tile)
    {
        dim3 grid(SEQ / 16, NHEADS);
        size_t shmem = (size_t)16 * SEQ * 4   // scores f32
                     + (size_t)16 * SEQ * 2   // P bf16
                     + 256 * 4                // histogram
                     + 16 * 4                 // thresholds
                     + 2 * 4;                 // select state
        const float scale = 0.08838834764831845f;  // 1/sqrt(128)
        attn_topk<<<grid, 256, shmem, stream>>>(Qb, Kb, Vtb, Cx, scale);
    }

    // 4) output projection (f32 out)
    {
        dim3 grid(HDIM / BN, SEQ / BM);
        gemm_bf16<<<grid, 256, 0, stream>>>(Cx, wob, bo, (float*)d_out,
                                            SEQ, HDIM, HDIM, 1);
    }
}
